// RIMCell_37692632990224
// MI455X (gfx1250) — matrix operand order repaired and checked by reading, not compiled, not run on hardware
//
#include <hip/hip_runtime.h>

// ---------------------------------------------------------------------------
// RIM cell forward for MI455X (gfx1250, wave32, WMMA).
// Dominant cost: 6-way batched GEMMs (gates via Wih/Whh ~ 29.5 GFLOP of 38 total)
// -> v_wmma_f32_16x16x32_bf16, f32 accumulate, double-buffered LDS staging.
// ---------------------------------------------------------------------------

typedef unsigned short u16;
typedef __bf16 bf16;
typedef bf16  v16bf __attribute__((ext_vector_type(16)));
typedef u16   v16u  __attribute__((ext_vector_type(16)));
typedef u16   v8u   __attribute__((ext_vector_type(8)));
typedef float v8f   __attribute__((ext_vector_type(8)));
typedef float f4    __attribute__((ext_vector_type(4)));

#define B_    1024
#define IN_   512
#define H_    600
#define NU_   6
#define TOPK_ 4
#define QK_   64
#define VD_   400
#define NCH_  4
#define CQ_   32
#define CK_   32
#define CV_   100
#define G4H_  2400   // 4*H

__device__ __forceinline__ u16 f2bf(float f) {
  union { float f; unsigned u; } v; v.f = f;
  unsigned r = 0x7FFFu + ((v.u >> 16) & 1u);   // round-to-nearest-even
  return (u16)((v.u + r) >> 16);
}
__device__ __forceinline__ float sigm(float x) { return 1.0f / (1.0f + expf(-x)); }

// ---------------------------------------------------------------------------
// f32 -> bf16 straight cast (Wih/Whh are already N x K for their einsum)
// ---------------------------------------------------------------------------
__global__ void k_cast(const float* __restrict__ src, u16* __restrict__ dst, long n) {
  long i = (long)blockIdx.x * blockDim.x + threadIdx.x;
  long stride = (long)gridDim.x * blockDim.x;
  for (; i < n; i += stride) dst[i] = f2bf(src[i]);
}

// f32 (u,K,N) -> bf16 (u,N,K): K becomes contiguous per output column.
__global__ void k_castT(const float* __restrict__ src, u16* __restrict__ dst,
                        int K, int N, long total) {
  long i = (long)blockIdx.x * blockDim.x + threadIdx.x;
  long stride = (long)gridDim.x * blockDim.x;
  long per = (long)K * N;
  for (; i < total; i += stride) {
    long u = i / per;
    long r = i - u * per;
    int  n = (int)(r / K);
    int  k = (int)(r - (long)n * K);
    dst[u * per + (long)n * K + k] = f2bf(src[u * per + (long)k * N + n]);
  }
}

__global__ void k_badd(const float* __restrict__ a, const float* __restrict__ b,
                       float* __restrict__ o, int n) {
  int i = blockIdx.x * blockDim.x + threadIdx.x;
  if (i < n) o[i] = a[i] + b[i];
}

// ---------------------------------------------------------------------------
// Batched GEMM:  C[u][m][n] (+)= sum_k A[u][m][k] * W[u][n][k] (+bias)
//   A: f32 (bf16-converted on the fly into LDS), W: bf16 bits, N x K row-major.
// Block: 256 threads (8 wave32); tile 64(M) x 128(N), K step 32, double-buffered.
// Waves 2(M) x 4(N); each wave owns 32x32 -> 4 v_wmma_f32_16x16x32_bf16 / step.
// Tile-uniform predication: vector loads (b128) except on the K tail / N edge.
// ---------------------------------------------------------------------------
__global__ __launch_bounds__(256) void k_gemm(
    const float* __restrict__ A, int lda, long sAu,
    const u16* __restrict__ W, long sWu,
    float* __restrict__ C, int ldc, long sCu,
    const float* __restrict__ bias, int sBu,
    int M, int N, int K, int accum)
{
  const int u = blockIdx.z;
  A += (long)u * sAu;
  W += (long)u * sWu;
  C += (long)u * sCu;

  const int m0 = blockIdx.y * 64;
  const int n0 = blockIdx.x * 128;
  const int tid  = threadIdx.x;
  const int lane = tid & 31;
  const int wave = tid >> 5;
  const int wm = wave >> 2;      // 0..1 (M strip of 32)
  const int wn = wave & 3;       // 0..3 (N strip of 32)

  __shared__ u16 As[2][64][40];    // 64 rows  x 32 k (+8 pad; 80B row, bank-clean)
  __shared__ u16 Bs[2][128][40];   // 128 cols x 32 k

  // A staging: thread -> (row, 8 consecutive k)
  const int trow = tid >> 2;            // 0..63
  const int tk   = (tid & 3) * 8;       // 0,8,16,24
  // B staging: thread -> (col, 16 consecutive k)
  const int tn = tid >> 1;              // 0..127
  const int wk = (tid & 1) * 16;        // 0,16

  const int gm = m0 + trow;
  const int gn = n0 + tn;
  const float* Arow = A + (long)gm * lda;
  const u16*   Wrow = W + (long)gn * K;
  const bool mOk = (gm < M);
  const bool nOk = (gn < N);

  const f4  fz = {0.f, 0.f, 0.f, 0.f};
  const v8u uz = {0, 0, 0, 0, 0, 0, 0, 0};

  f4  ra0 = fz, ra1 = fz;
  v8u rb0 = uz, rb1 = uz;

  // stage next tile (global -> registers); tile-uniform K-tail predication
  auto stage = [&](int k0) {
    if (mOk) {
      if (k0 + 32 <= K) {
        ra0 = *(const f4*)(Arow + k0 + tk);
        ra1 = *(const f4*)(Arow + k0 + tk + 4);
      } else {
        #pragma unroll
        for (int j = 0; j < 4; ++j) {
          int kk = k0 + tk + j;
          ra0[j] = (kk     < K) ? Arow[kk]     : 0.f;
          ra1[j] = (kk + 4 < K) ? Arow[kk + 4] : 0.f;
        }
      }
    } else { ra0 = fz; ra1 = fz; }
    if (nOk) {
      if (k0 + 32 <= K) {
        rb0 = *(const v8u*)(Wrow + k0 + wk);
        rb1 = *(const v8u*)(Wrow + k0 + wk + 8);
        if (k0 + 64 <= K)
          __builtin_prefetch(Wrow + k0 + 32 + wk, 0, 1);   // global_prefetch_b8
      } else {
        #pragma unroll
        for (int j = 0; j < 8; ++j) {
          int kk = k0 + wk + j;
          rb0[j] = (kk     < K) ? Wrow[kk]     : (u16)0;
          rb1[j] = (kk + 8 < K) ? Wrow[kk + 8] : (u16)0;
        }
      }
    } else { rb0 = uz; rb1 = uz; }
  };

  // commit staged registers to LDS buffer (b128 stores)
  auto commit = [&](int buf) {
    v8u pa;
    #pragma unroll
    for (int j = 0; j < 4; ++j) {
      pa[j]     = f2bf(ra0[j]);
      pa[j + 4] = f2bf(ra1[j]);
    }
    *(v8u*)&As[buf][trow][tk]    = pa;
    *(v8u*)&Bs[buf][tn][wk]      = rb0;
    *(v8u*)&Bs[buf][tn][wk + 8]  = rb1;
  };

  v8f acc[2][2] = {};

  const int rA = wm * 32 + (lane & 15);
  const int cB = wn * 32 + (lane & 15);
  const int ka = (lane < 16) ? 0 : 16;   // A frag: lanes 16-31 take K=16..31
  const int kb = (lane < 16) ? 0 : 16;   // B frag: lanes 16-31 take K=16..31

  auto compute = [&](int buf) {
    v16u af0, af1;
    #pragma unroll
    for (int j = 0; j < 8; ++j) {
      af0[j]     = As[buf][rA][ka + j];
      af0[j + 8] = As[buf][rA][ka + 8 + j];
      af1[j]     = As[buf][rA + 16][ka + j];
      af1[j + 8] = As[buf][rA + 16][ka + 8 + j];
    }
    v16u bf0, bf1;
    #pragma unroll
    for (int j = 0; j < 16; ++j) {
      bf0[j] = Bs[buf][cB][kb + j];
      bf1[j] = Bs[buf][cB + 16][kb + j];
    }
    acc[0][0] = __builtin_amdgcn_wmma_f32_16x16x32_bf16(
        false, __builtin_bit_cast(v16bf, af0), false, __builtin_bit_cast(v16bf, bf0),
        (short)0, acc[0][0], false, false);
    acc[0][1] = __builtin_amdgcn_wmma_f32_16x16x32_bf16(
        false, __builtin_bit_cast(v16bf, af0), false, __builtin_bit_cast(v16bf, bf1),
        (short)0, acc[0][1], false, false);
    acc[1][0] = __builtin_amdgcn_wmma_f32_16x16x32_bf16(
        false, __builtin_bit_cast(v16bf, af1), false, __builtin_bit_cast(v16bf, bf0),
        (short)0, acc[1][0], false, false);
    acc[1][1] = __builtin_amdgcn_wmma_f32_16x16x32_bf16(
        false, __builtin_bit_cast(v16bf, af1), false, __builtin_bit_cast(v16bf, bf1),
        (short)0, acc[1][1], false, false);
  };

  const int nk = (K + 31) >> 5;
  stage(0);
  commit(0);
  __syncthreads();
  for (int t = 0; t < nk; ++t) {
    const int cur = t & 1;
    const bool have = (t + 1) < nk;
    if (have) stage((t + 1) << 5);   // global loads fly under the WMMAs
    compute(cur);
    if (have) commit(cur ^ 1);
    __syncthreads();
  }

  // --- store: C/D layout = VGPR i -> row (lane<16 ? i : i+8), col = lane&15 ---
  #pragma unroll
  for (int ni = 0; ni < 2; ++ni) {
    const int col = n0 + wn * 32 + ni * 16 + (lane & 15);
    if (col >= N) continue;
    const float bval = bias ? bias[(long)u * sBu + col] : 0.f;
    #pragma unroll
    for (int mi = 0; mi < 2; ++mi) {
      #pragma unroll
      for (int i = 0; i < 8; ++i) {
        int row = m0 + wm * 32 + mi * 16 + ((lane < 16) ? i : i + 8);
        if (row >= M) continue;
        float v = acc[mi][ni][i] + bval;
        long idx = (long)row * ldc + col;
        if (accum) v += C[idx];
        C[idx] = v;
      }
    }
  }
}

// ---------------------------------------------------------------------------
// scores (b,u,t) = <ql[b,u], kl[b,t]>/8; t=0: x@Wk+bk, t=1: bk (zero token).
// top-4 of scores[:,0] -> mask; 2-way softmax -> probs. One wave per b.
// ---------------------------------------------------------------------------
__global__ __launch_bounds__(32) void k_scores(
    const float* __restrict__ ql, const float* __restrict__ kl0,
    const float* __restrict__ bk, float* __restrict__ probs,
    float* __restrict__ maskb)
{
  int b = blockIdx.x;
  int lane = threadIdx.x;
  __shared__ float s[NU_][2];
  if (lane < NU_ * 2) {
    int u = lane >> 1, t = lane & 1;
    const float* q  = ql + ((long)b * NU_ + u) * QK_;
    const float* kv = (t == 0) ? (kl0 + (long)b * QK_) : bk;
    float acc = 0.f;
    #pragma unroll 8
    for (int i = 0; i < QK_; ++i) acc += q[i] * kv[i];
    s[u][t] = acc * 0.125f;   // 1/sqrt(64)
  }
  __syncthreads();
  if (lane == 0) {
    int chosen[NU_] = {0, 0, 0, 0, 0, 0};
    for (int kk = 0; kk < TOPK_; ++kk) {          // stable argmax (ties -> low idx)
      int best = 0; float bv = -3.4e38f;
      for (int uu = 0; uu < NU_; ++uu)
        if (!chosen[uu] && s[uu][0] > bv) { bv = s[uu][0]; best = uu; }
      chosen[best] = 1;
    }
    for (int uu = 0; uu < NU_; ++uu) {
      maskb[(long)b * NU_ + uu] = chosen[uu] ? 1.f : 0.f;
      float s0 = s[uu][0], s1 = s[uu][1];
      float mx = fmaxf(s0, s1);
      float e0 = expf(s0 - mx), e1 = expf(s1 - mx);
      float inv = 1.f / (e0 + e1);
      probs[((long)b * NU_ + uu) * 2 + 0] = e0 * inv;
      probs[((long)b * NU_ + uu) * 2 + 1] = e1 * inv;
    }
  }
}

// inputs[b,u,v] = (p0*vl0[b,v] + p1*bv[v]) * mask[b,u]
__global__ void k_inputs(const float* __restrict__ probs, const float* __restrict__ maskb,
                         const float* __restrict__ vl0, const float* __restrict__ bv,
                         float* __restrict__ inputs, long n)
{
  long i = (long)blockIdx.x * blockDim.x + threadIdx.x;
  long stride = (long)gridDim.x * blockDim.x;
  for (; i < n; i += stride) {
    long bu = i / VD_;
    int  v  = (int)(i - bu * VD_);
    long b  = bu / NU_;
    float p0 = probs[bu * 2 + 0], p1 = probs[bu * 2 + 1];
    inputs[i] = (p0 * vl0[b * VD_ + v] + p1 * bv[v]) * maskb[bu];
  }
}

// LSTM pointwise: gates layout (b,u, [i|f|g|o] * H)
__global__ void k_lstm(const float* __restrict__ gates, const float* __restrict__ cs,
                       float* __restrict__ cnew, float* __restrict__ hrnn, long n)
{
  long i = (long)blockIdx.x * blockDim.x + threadIdx.x;
  long stride = (long)gridDim.x * blockDim.x;
  for (; i < n; i += stride) {
    long bu = i / H_;
    int  h  = (int)(i - bu * H_);
    long gb = bu * G4H_;
    float ig = gates[gb + h];
    float fg = gates[gb + H_ + h];
    float gg = gates[gb + 2 * H_ + h];
    float og = gates[gb + 3 * H_ + h];
    float c  = sigm(fg) * cs[i] + sigm(ig) * tanhf(gg);
    cnew[i] = c;
    hrnn[i] = sigm(og) * tanhf(c);
  }
}

// ---------------------------------------------------------------------------
// Communication attention: one wave per (b, head). 6x6 scores over CK=32,
// softmax over t, mask on query unit, ctx[b,u,ch*CV+v] = sum_t att * vc.
// ---------------------------------------------------------------------------
__global__ __launch_bounds__(32) void k_comm(
    const float* __restrict__ qc, const float* __restrict__ kc,
    const float* __restrict__ vc, const float* __restrict__ maskb,
    float* __restrict__ ctx)
{
  int b  = blockIdx.x / NCH_;
  int ch = blockIdx.x % NCH_;
  int lane = threadIdx.x;
  __shared__ float att[NU_][NU_];

  #pragma unroll
  for (int ut = 0; ut < NU_ * NU_; ++ut) {
    int u = ut / NU_, t = ut % NU_;
    float p = qc[((long)b * NU_ + u) * (NCH_ * CQ_) + ch * CQ_ + lane] *
              kc[((long)b * NU_ + t) * (NCH_ * CK_) + ch * CK_ + lane];
    #pragma unroll
    for (int off = 16; off; off >>= 1) p += __shfl_xor(p, off, 32);
    if (lane == 0) att[u][t] = p * 0.17677669529663687f;   // 1/sqrt(32)
  }
  __syncthreads();
  if (lane < NU_) {
    float mx = -3.4e38f;
    for (int t = 0; t < NU_; ++t) mx = fmaxf(mx, att[lane][t]);
    float e[NU_]; float ssum = 0.f;
    for (int t = 0; t < NU_; ++t) { e[t] = expf(att[lane][t] - mx); ssum += e[t]; }
    float inv = maskb[(long)b * NU_ + lane] / ssum;
    for (int t = 0; t < NU_; ++t) att[lane][t] = e[t] * inv;
  }
  __syncthreads();
  for (int u = 0; u < NU_; ++u)
    for (int v = lane; v < CV_; v += 32) {
      float acc = 0.f;
      #pragma unroll
      for (int t = 0; t < NU_; ++t)
        acc += att[u][t] * vc[((long)b * NU_ + t) * (NCH_ * CV_) + ch * CV_ + v];
      ctx[((long)b * NU_ + u) * (NCH_ * CV_) + ch * CV_ + v] = acc;
    }
}

// ---------------------------------------------------------------------------
// y = ctx2 + h_rnn; layernorm(y); blend with mask into hs_out/cs_out.
// One wave per (b,u); 8 waves per block.
// ---------------------------------------------------------------------------
__global__ __launch_bounds__(256) void k_ln_out(
    const float* __restrict__ ctx2, const float* __restrict__ hrnn,
    const float* __restrict__ cnew, const float* __restrict__ hs,
    const float* __restrict__ cs, const float* __restrict__ maskb,
    const float* __restrict__ ln_g, const float* __restrict__ ln_b,
    float* __restrict__ hs_out, float* __restrict__ cs_out, int total)
{
  int pair = blockIdx.x * 8 + (threadIdx.x >> 5);
  if (pair >= total) return;
  int lane = threadIdx.x & 31;
  long base = (long)pair * H_;

  float yv[19];
  float sum = 0.f;
  for (int h = lane, i = 0; h < H_; h += 32, ++i) {
    float y = ctx2[base + h] + hrnn[base + h];
    yv[i] = y;
    sum += y;
  }
  #pragma unroll
  for (int off = 16; off; off >>= 1) sum += __shfl_xor(sum, off, 32);
  float mu = sum * (1.0f / H_);

  float vs = 0.f;
  for (int h = lane, i = 0; h < H_; h += 32, ++i) {
    float d = yv[i] - mu;
    vs += d * d;
  }
  #pragma unroll
  for (int off = 16; off; off >>= 1) vs += __shfl_xor(vs, off, 32);
  float rstd = rsqrtf(vs * (1.0f / H_) + 1e-5f);

  float m = maskb[pair];
  for (int h = lane, i = 0; h < H_; h += 32, ++i) {
    float hn = (yv[i] - mu) * rstd * ln_g[h] + ln_b[h];
    hs_out[base + h] = m * hn + (1.f - m) * hs[base + h];
    cs_out[base + h] = m * cnew[base + h] + (1.f - m) * cs[base + h];
  }
}

// ---------------------------------------------------------------------------
// Host-side orchestration
// ---------------------------------------------------------------------------
extern "C" void kernel_launch(void* const* d_in, const int* in_sizes, int n_in,
                              void* d_out, int out_size, void* d_ws, size_t ws_size,
                              hipStream_t stream) {
  const float* x    = (const float*)d_in[0];
  const float* hs   = (const float*)d_in[1];
  const float* cs   = (const float*)d_in[2];
  const float* Wk   = (const float*)d_in[3];
  const float* bk   = (const float*)d_in[4];
  const float* Wv   = (const float*)d_in[5];
  const float* bv   = (const float*)d_in[6];
  const float* Wq   = (const float*)d_in[7];
  const float* Wqc  = (const float*)d_in[8];
  const float* Wkc  = (const float*)d_in[9];
  const float* Wvc  = (const float*)d_in[10];
  const float* Woc  = (const float*)d_in[11];
  const float* Wih  = (const float*)d_in[12];
  const float* Whh  = (const float*)d_in[13];
  const float* bih  = (const float*)d_in[14];
  const float* bhh  = (const float*)d_in[15];
  const float* ln_g = (const float*)d_in[16];
  const float* ln_b = (const float*)d_in[17];

  float* hs_out = (float*)d_out;
  float* cs_out = hs_out + (long)B_ * NU_ * H_;

  // ---- workspace carve-up ----
  size_t off = 0;
  char* base = (char*)d_ws;
  auto alloc = [&](size_t bytes) -> char* {
    char* p = base + off;
    off += (bytes + 255) & ~(size_t)255;
    return p;
  };
  u16* wkT   = (u16*)alloc((size_t)QK_ * IN_ * 2);
  u16* wvT   = (u16*)alloc((size_t)VD_ * IN_ * 2);
  u16* wqT   = (u16*)alloc((size_t)NU_ * QK_ * H_ * 2);
  u16* wqcT  = (u16*)alloc((size_t)NU_ * NCH_ * CQ_ * H_ * 2);
  u16* wkcT  = (u16*)alloc((size_t)NU_ * NCH_ * CK_ * H_ * 2);
  u16* wvcT  = (u16*)alloc((size_t)NU_ * NCH_ * CV_ * H_ * 2);
  u16* wocT  = (u16*)alloc((size_t)NU_ * H_ * VD_ * 2);
  u16* wihB  = (u16*)alloc((size_t)NU_ * G4H_ * VD_ * 2);
  u16* whhB  = (u16*)alloc((size_t)NU_ * G4H_ * H_ * 2);
  float* biassum = (float*)alloc((size_t)NU_ * G4H_ * 4);
  float* kl0   = (float*)alloc((size_t)B_ * QK_ * 4);
  float* vl0   = (float*)alloc((size_t)B_ * VD_ * 4);
  float* ql    = (float*)alloc((size_t)B_ * NU_ * QK_ * 4);
  float* probs = (float*)alloc((size_t)B_ * NU_ * 2 * 4);
  float* maskb = (float*)alloc((size_t)B_ * NU_ * 4);
  float* inputs= (float*)alloc((size_t)B_ * NU_ * VD_ * 4);
  float* gates = (float*)alloc((size_t)B_ * NU_ * G4H_ * 4);   // 59 MB, recycled below
  float* cnew  = (float*)alloc((size_t)B_ * NU_ * H_ * 4);
  float* hrnn  = (float*)alloc((size_t)B_ * NU_ * H_ * 4);
  // gates region reuse (after k_lstm consumed it): qc/kc/vc/ctx/ctx2 = 40.7 MB < 59 MB
  float* qc   = gates;
  float* kcb  = qc  + (size_t)B_ * NU_ * NCH_ * CQ_;
  float* vcb  = kcb + (size_t)B_ * NU_ * NCH_ * CK_;
  float* ctx  = vcb + (size_t)B_ * NU_ * NCH_ * CV_;
  float* ctx2 = ctx + (size_t)B_ * NU_ * NCH_ * CV_;

  auto gsb = [](long n) -> unsigned {                 // grid-stride block count
    long b = (n + 2047) / 2048;
    if (b < 1) b = 1;
    if (b > 65535) b = 65535;
    return (unsigned)b;
  };
  dim3 blk256(256);

  // ---- phase 0: weight casts (bf16, N x K layout) ----
  k_cast <<<gsb((long)NU_*G4H_*VD_), blk256, 0, stream>>>(Wih, wihB, (long)NU_*G4H_*VD_);
  k_cast <<<gsb((long)NU_*G4H_*H_ ), blk256, 0, stream>>>(Whh, whhB, (long)NU_*G4H_*H_);
  k_castT<<<gsb((long)IN_*QK_),          blk256, 0, stream>>>(Wk,  wkT,  IN_, QK_,        (long)IN_*QK_);
  k_castT<<<gsb((long)IN_*VD_),          blk256, 0, stream>>>(Wv,  wvT,  IN_, VD_,        (long)IN_*VD_);
  k_castT<<<gsb((long)NU_*H_*QK_),       blk256, 0, stream>>>(Wq,  wqT,  H_,  QK_,        (long)NU_*H_*QK_);
  k_castT<<<gsb((long)NU_*H_*NCH_*CQ_),  blk256, 0, stream>>>(Wqc, wqcT, H_,  NCH_*CQ_,   (long)NU_*H_*NCH_*CQ_);
  k_castT<<<gsb((long)NU_*H_*NCH_*CK_),  blk256, 0, stream>>>(Wkc, wkcT, H_,  NCH_*CK_,   (long)NU_*H_*NCH_*CK_);
  k_castT<<<gsb((long)NU_*H_*NCH_*CV_),  blk256, 0, stream>>>(Wvc, wvcT, H_,  NCH_*CV_,   (long)NU_*H_*NCH_*CV_);
  k_castT<<<gsb((long)NU_*VD_*H_),       blk256, 0, stream>>>(Woc, wocT, VD_, H_,         (long)NU_*VD_*H_);
  k_badd <<<(NU_*G4H_ + 255)/256, blk256, 0, stream>>>(bih, bhh, biassum, NU_*G4H_);

  const int MT = B_ / 64;   // 16 M-tiles
  #define NT(n) (((n) + 127) / 128)

  // ---- phase 1: input projections ----
  // kl0 = x @ Wk + bk            (1024x64, K=512)
  k_gemm<<<dim3(NT(QK_), MT, 1), blk256, 0, stream>>>(x, IN_, 0, wkT, 0,
      kl0, QK_, 0, bk, 0, B_, QK_, IN_, 0);
  // vl0 = x @ Wv + bv            (1024x400, K=512)
  k_gemm<<<dim3(NT(VD_), MT, 1), blk256, 0, stream>>>(x, IN_, 0, wvT, 0,
      vl0, VD_, 0, bv, 0, B_, VD_, IN_, 0);
  // ql[u] = hs[:,u,:] @ Wq[u]    (1024x64, K=600) x6
  k_gemm<<<dim3(NT(QK_), MT, NU_), blk256, 0, stream>>>(hs, NU_*H_, H_, wqT, (long)QK_*H_,
      ql, NU_*QK_, QK_, nullptr, 0, B_, QK_, H_, 0);

  // ---- phase 2: scores, top-k mask, softmax, input mixing ----
  k_scores<<<B_, 32, 0, stream>>>(ql, kl0, bk, probs, maskb);
  k_inputs<<<gsb((long)B_*NU_*VD_), blk256, 0, stream>>>(probs, maskb, vl0, bv,
      inputs, (long)B_*NU_*VD_);

  // ---- phase 3: LSTM gates (the two big WMMA GEMMs) ----
  // gates  = inputs @ Wih[u]^T + (bih+bhh)   (1024x2400, K=400) x6
  k_gemm<<<dim3(NT(G4H_), MT, NU_), blk256, 0, stream>>>(inputs, NU_*VD_, VD_,
      wihB, (long)G4H_*VD_, gates, NU_*G4H_, G4H_, biassum, G4H_, B_, G4H_, VD_, 0);
  // gates += hs @ Whh[u]^T                   (1024x2400, K=600) x6
  k_gemm<<<dim3(NT(G4H_), MT, NU_), blk256, 0, stream>>>(hs, NU_*H_, H_,
      whhB, (long)G4H_*H_, gates, NU_*G4H_, G4H_, nullptr, 0, B_, G4H_, H_, 1);
  k_lstm<<<gsb((long)B_*NU_*H_), blk256, 0, stream>>>(gates, cs, cnew, hrnn,
      (long)B_*NU_*H_);

  // ---- phase 4: communication attention (gates region recycled from here) ----
  k_gemm<<<dim3(NT(NCH_*CQ_), MT, NU_), blk256, 0, stream>>>(hrnn, NU_*H_, H_,
      wqcT, (long)NCH_*CQ_*H_, qc, NU_*NCH_*CQ_, NCH_*CQ_, nullptr, 0, B_, NCH_*CQ_, H_, 0);
  k_gemm<<<dim3(NT(NCH_*CK_), MT, NU_), blk256, 0, stream>>>(hrnn, NU_*H_, H_,
      wkcT, (long)NCH_*CK_*H_, kcb, NU_*NCH_*CK_, NCH_*CK_, nullptr, 0, B_, NCH_*CK_, H_, 0);
  k_gemm<<<dim3(NT(NCH_*CV_), MT, NU_), blk256, 0, stream>>>(hrnn, NU_*H_, H_,
      wvcT, (long)NCH_*CV_*H_, vcb, NU_*NCH_*CV_, NCH_*CV_, nullptr, 0, B_, NCH_*CV_, H_, 0);
  k_comm<<<B_*NCH_, 32, 0, stream>>>(qc, kcb, vcb, maskb, ctx);
  // ctx2 = ctx @ Woc[u]^T                    (1024x600, K=400) x6
  k_gemm<<<dim3(NT(H_), MT, NU_), blk256, 0, stream>>>(ctx, NU_*NCH_*CV_, NCH_*CV_,
      wocT, (long)H_*VD_, ctx2, NU_*H_, H_, nullptr, 0, B_, H_, VD_, 0);

  // ---- phase 5: layernorm + masked state update ----
  k_ln_out<<<(B_*NU_ + 7)/8, blk256, 0, stream>>>(ctx2, hrnn, cnew, hs, cs, maskb,
      ln_g, ln_b, hs_out, cs_out, B_*NU_);

  (void)in_sizes; (void)n_in; (void)out_size; (void)ws_size;
}
